// KeypointMHA_64098091925554
// MI455X (gfx1250) — matrix operand order repaired and checked by reading, not compiled, not run on hardware
//
#include <hip/hip_runtime.h>
#include <hip/hip_bf16.h>

// ---------------------------------------------------------------------------
// KeypointMHA on MI455X (gfx1250): all matmuls via v_wmma_f32_16x16x32_f16,
// flash-attention single pass with 64-atom steps (16 WMMAs / iteration).
// The QK scale (incl. log2e for exp2-domain softmax) is folded into the Q
// projection, so the attention loop pays one FMAC per score element.
// ---------------------------------------------------------------------------

typedef __attribute__((ext_vector_type(16))) _Float16 v16h;
typedef __attribute__((ext_vector_type(8)))  _Float16 v8h;
typedef __attribute__((ext_vector_type(8)))  float    v8f;

#define NHEADS 16
#define HID    64
#define IND    256
#define OUTD   256
#define NKP    1024
#define NATM   8192
#define HD     1024   // NHEADS*HID

static __device__ __forceinline__ v8f wmma(v16h a, v16h b, v8f c) {
  // (neg_a, A, neg_b, B, c_mod, C, reuse_a, reuse_b)
  return __builtin_amdgcn_wmma_f32_16x16x32_f16(false, a, false, b, (short)0, c,
                                                false, false);
}

// A-operand (16x32 f16): lane holds row m=lane%16; K elems 2*off..2*off+15
// (contiguous) with off=(lane/16)*8.  Source row-major f16.
static __device__ __forceinline__ v16h loadA_h(const _Float16* row, int kchunk, int off) {
  v8h lo = *(const v8h*)(row + kchunk + 2 * off);
  v8h hi = *(const v8h*)(row + kchunk + 2 * off + 8);
  v16h a;
#pragma unroll
  for (int j = 0; j < 8; ++j) { a[j] = lo[j]; a[j + 8] = hi[j]; }
  return a;
}

// Same A layout but converting from f32 source.
static __device__ __forceinline__ v16h loadA_f(const float* row, int kchunk, int off) {
  v16h a;
#pragma unroll
  for (int j = 0; j < 8; ++j) {
    a[j]     = (_Float16)row[kchunk + 2 * off + j];
    a[j + 8] = (_Float16)row[kchunk + 2 * off + 8 + j];
  }
  return a;
}

// B-operand (32x16 f16): lane holds col n=lane%16, K elems (lane/16)*16 + j,
// contiguous -> 16 consecutive f32 converted.
static __device__ __forceinline__ v16h loadB_f(const float* p) {
  v16h b;
#pragma unroll
  for (int j = 0; j < 16; ++j) b[j] = (_Float16)p[j];
  return b;
}

// ---------------------------------------------------------------------------
// Projection GEMM: Y = (X[M,256] @ W[1024,256]^T) * oscale -> f16.  Each wave
// computes a 16x64 output tile (one A fragment shared by 4 WMMA column tiles).
// MODE 0: row-major [M,1024].  MODE 1 (V): store transposed per head:
//   Vt[(h*64+d)*8192 + a'],  h=a>>9, a'=((a&511)<<4)|(c>>6), d=c&63.
// ---------------------------------------------------------------------------
template <int MODE>
__global__ __launch_bounds__(32) void proj_gemm(const float* __restrict__ X,
                                                const float* __restrict__ W,
                                                _Float16* __restrict__ Y,
                                                float oscale) {
  const int lane  = threadIdx.x;
  const int mtile = blockIdx.x >> 4;       // HD/64 = 16 column blocks
  const int nblk  = blockIdx.x & 15;
  const int r16 = lane & 15, hg = lane >> 4, off = hg * 8;

  const float* xrow = X + (mtile * 16 + r16) * IND;
  const float* wrow0 = W + (nblk * 64 +  0 + r16) * IND + hg * 16;
  const float* wrow1 = W + (nblk * 64 + 16 + r16) * IND + hg * 16;
  const float* wrow2 = W + (nblk * 64 + 32 + r16) * IND + hg * 16;
  const float* wrow3 = W + (nblk * 64 + 48 + r16) * IND + hg * 16;

  v8f acc[4] = {};
#pragma unroll
  for (int kb = 0; kb < IND; kb += 32) {
    const v16h a = loadA_f(xrow, kb, off);
    acc[0] = wmma(a, loadB_f(wrow0 + kb), acc[0]);
    acc[1] = wmma(a, loadB_f(wrow1 + kb), acc[1]);
    acc[2] = wmma(a, loadB_f(wrow2 + kb), acc[2]);
    acc[3] = wmma(a, loadB_f(wrow3 + kb), acc[3]);
  }
#pragma unroll
  for (int t = 0; t < 4; ++t) {
#pragma unroll
    for (int r = 0; r < 8; ++r) {
      const int m = mtile * 16 + hg * 8 + r;  // C layout: lanes16-31 hold M=8..15
      const int n = nblk * 64 + t * 16 + r16;
      const float v = acc[t][r] * oscale;
      if constexpr (MODE == 0) {
        Y[m * HD + n] = (_Float16)v;
      } else {
        const int h = m >> 9, a2 = ((m & 511) << 4) | (n >> 6), d = n & 63;
        Y[(h * HID + d) * NATM + a2] = (_Float16)v;
      }
    }
  }
}

// ---------------------------------------------------------------------------
// Flash attention: one wave per (head, 16-query tile), 64-atom steps.
// Q was pre-scaled by log2e/(8*sqrt2); bias folded with one FMAC per element.
// ---------------------------------------------------------------------------
__global__ __launch_bounds__(32) void attn_kernel(const _Float16* __restrict__ Qb,
                                                  const _Float16* __restrict__ Kb,
                                                  const _Float16* __restrict__ Vtb,
                                                  const float* __restrict__ dist,
                                                  _Float16* __restrict__ kpw) {
  __shared__ _Float16 P[16 * 64];   // prob tile staged C-layout -> A-layout

  const int lane = threadIdx.x;
  const int h = blockIdx.x & 15;
  const int qbase = (blockIdx.x >> 4) * 16;
  const int r16 = lane & 15, hg = lane >> 4, off = hg * 8;

  const _Float16* Qh = Qb  + h * (64 * 1024);    // [1024,64] head slab
  const _Float16* Kh = Kb  + h * (512 * 1024);   // [8192,64]
  const _Float16* Vt = Vtb + h * (64 * NATM);    // [64,8192] pre-transposed

  const float SC_D = 0.4082482904638631f * 1.44269504088896f;  // log2e/(sqrt6)

  const _Float16* qrow = Qh + (qbase + r16) * HID;
  const v16h aQ0 = loadA_h(qrow, 0, off);
  const v16h aQ1 = loadA_h(qrow, 32, off);

  float mi[8], li[8];
  v8f o[4] = {};
#pragma unroll
  for (int r = 0; r < 8; ++r) { mi[r] = -1e30f; li[r] = 0.0f; }

  const float* drow = dist + (qbase + hg * 8) * NATM;

  for (int ab = 0; ab < NATM; ab += 64) {
    if (ab + 256 < NATM) {
      __builtin_prefetch((const void*)(Kh + (ab + 256 + r16) * HID), 0, 1);
      __builtin_prefetch((const void*)(Vt + r16 * NATM + ab + 256), 0, 1);
    }

    // ---- S = Q K^T : 4 column tiles x (K-dim 64 as two WMMAs) ----
    v8f s[4] = {};
#pragma unroll
    for (int j = 0; j < 4; ++j) {
      const _Float16* kc = Kh + (ab + 16 * j + r16) * HID + hg * 16;
      s[j] = wmma(aQ0, *(const v16h*)kc,        s[j]);
      s[j] = wmma(aQ1, *(const v16h*)(kc + 32), s[j]);
    }

    // ---- add scaled bias (one FMAC/elem), row max over 64 cols ----
    float rm[8];
#pragma unroll
    for (int r = 0; r < 8; ++r) {
#pragma unroll
      for (int j = 0; j < 4; ++j) {
        const float b = drow[r * NATM + ab + 16 * j + r16];
        s[j][r] = fmaf(b, SC_D, s[j][r]);
      }
      rm[r] = fmaxf(fmaxf(s[0][r], s[1][r]), fmaxf(s[2][r], s[3][r]));
    }
#pragma unroll
    for (int r = 0; r < 8; ++r)
#pragma unroll
      for (int mk = 1; mk < 16; mk <<= 1)
        rm[r] = fmaxf(rm[r], __shfl_xor(rm[r], mk, 32));

    // ---- online softmax update (exp2) ----
    float rs[8];
#pragma unroll
    for (int r = 0; r < 8; ++r) {
      const float mn = fmaxf(mi[r], rm[r]);
      const float al = __builtin_amdgcn_exp2f(mi[r] - mn);
      mi[r] = mn;
      float acc = 0.0f;
#pragma unroll
      for (int j = 0; j < 4; ++j) {
        const float p = __builtin_amdgcn_exp2f(s[j][r] - mn);
        s[j][r] = p;
        acc += p;
      }
      rs[r] = acc;
      li[r] *= al;
      o[0][r] *= al; o[1][r] *= al; o[2][r] *= al; o[3][r] *= al;
    }
#pragma unroll
    for (int r = 0; r < 8; ++r) {
#pragma unroll
      for (int mk = 1; mk < 16; mk <<= 1) rs[r] += __shfl_xor(rs[r], mk, 32);
      li[r] += rs[r];
    }

    // ---- stage P (C layout) -> LDS -> reload in A layout ----
#pragma unroll
    for (int r = 0; r < 8; ++r) {
      const int base = (hg * 8 + r) * 64;
      P[base +      r16] = (_Float16)s[0][r];
      P[base + 16 + r16] = (_Float16)s[1][r];
      P[base + 32 + r16] = (_Float16)s[2][r];
      P[base + 48 + r16] = (_Float16)s[3][r];
    }
    __syncthreads();
    const v16h aP0 = loadA_h(P + r16 * 64, 0, off);
    const v16h aP1 = loadA_h(P + r16 * 64, 32, off);

    // ---- O += P · V : 4 d-chunks x (K-dim 64 as two WMMAs) ----
#pragma unroll
    for (int dc = 0; dc < 4; ++dc) {
      const _Float16* vrow = Vt + (dc * 16 + r16) * NATM + ab + hg * 16;
      o[dc] = wmma(aP0, *(const v16h*)vrow,        o[dc]);
      o[dc] = wmma(aP1, *(const v16h*)(vrow + 32), o[dc]);
    }
    __syncthreads();
  }

  // ---- normalize, write kp_wise[k', h*64 + d] as f16 ----
#pragma unroll
  for (int r = 0; r < 8; ++r) {
    const float inv = 1.0f / li[r];
    _Float16* dst = kpw + (qbase + hg * 8 + r) * HD + h * HID;
#pragma unroll
    for (int dc = 0; dc < 4; ++dc)
      dst[dc * 16 + r16] = (_Float16)(o[dc][r] * inv);
  }
}

// ---------------------------------------------------------------------------
// out = kp_wise[1024,1024](f16) @ Wo[256,1024]^T -> f32 [1024,256]
// Each wave computes a 16x64 output tile.
// ---------------------------------------------------------------------------
__global__ __launch_bounds__(32) void out_gemm(const _Float16* __restrict__ kpw,
                                               const float* __restrict__ Wo,
                                               float* __restrict__ out) {
  const int lane  = threadIdx.x;
  const int mtile = blockIdx.x >> 2;       // OUTD/64 = 4 column blocks
  const int nblk  = blockIdx.x & 3;
  const int r16 = lane & 15, hg = lane >> 4, off = hg * 8;

  const _Float16* arow = kpw + (mtile * 16 + r16) * HD;
  const float* wrow0 = Wo + (nblk * 64 +  0 + r16) * HD + hg * 16;
  const float* wrow1 = Wo + (nblk * 64 + 16 + r16) * HD + hg * 16;
  const float* wrow2 = Wo + (nblk * 64 + 32 + r16) * HD + hg * 16;
  const float* wrow3 = Wo + (nblk * 64 + 48 + r16) * HD + hg * 16;

  v8f acc[4] = {};
#pragma unroll 4
  for (int kb = 0; kb < HD; kb += 32) {
    const v16h a = loadA_h(arow, kb, off);
    acc[0] = wmma(a, loadB_f(wrow0 + kb), acc[0]);
    acc[1] = wmma(a, loadB_f(wrow1 + kb), acc[1]);
    acc[2] = wmma(a, loadB_f(wrow2 + kb), acc[2]);
    acc[3] = wmma(a, loadB_f(wrow3 + kb), acc[3]);
  }
#pragma unroll
  for (int t = 0; t < 4; ++t)
#pragma unroll
    for (int r = 0; r < 8; ++r)
      out[(mtile * 16 + hg * 8 + r) * OUTD + nblk * 64 + t * 16 + r16] = acc[t][r];
}

// ---------------------------------------------------------------------------
extern "C" void kernel_launch(void* const* d_in, const int* in_sizes, int n_in,
                              void* d_out, int out_size, void* d_ws, size_t ws_size,
                              hipStream_t stream) {
  const float* kp_feat = (const float*)d_in[0];
  const float* kp_dist = (const float*)d_in[1];
  const float* rec     = (const float*)d_in[2];
  const float* Wk      = (const float*)d_in[3];
  const float* Wq      = (const float*)d_in[4];
  const float* Wv      = (const float*)d_in[5];
  const float* Wo      = (const float*)d_in[6];

  // Workspace (f16): Q 2MB | K 16MB | Vt 16MB | kp_wise 2MB  = 36MB total.
  _Float16* Qb  = (_Float16*)d_ws;
  _Float16* Kb  = Qb + (size_t)NKP * HD;
  _Float16* Vt  = Kb + (size_t)NATM * HD;
  _Float16* kpw = Vt + (size_t)NATM * HD;

  // Fold the whole QK logit scale (1/(8*sqrt2)) and the exp2-domain log2e
  // into the Q projection output.
  const float q_scale = 0.08838834764831845f * 1.44269504088896f;

  proj_gemm<0><<<(NKP  / 16) * (HD / 64), 32, 0, stream>>>(kp_feat, Wq, Qb, q_scale);
  proj_gemm<0><<<(NATM / 16) * (HD / 64), 32, 0, stream>>>(rec,     Wk, Kb, 1.0f);
  proj_gemm<1><<<(NATM / 16) * (HD / 64), 32, 0, stream>>>(rec,     Wv, Vt, 1.0f);

  attn_kernel<<<NHEADS * (NKP / 16), 32, 0, stream>>>(Qb, Kb, Vt, kp_dist, kpw);

  out_gemm<<<(NKP / 16) * (OUTD / 64), 32, 0, stream>>>(kpw, Wo, (float*)d_out);
}
